// GuidanceRefinementBlock_35235911696699
// MI455X (gfx1250) — matrix operand order repaired and checked by reading, not compiled, not run on hardware
//
#include <hip/hip_runtime.h>
#include <hip/hip_bf16.h>

typedef __attribute__((ext_vector_type(16))) __bf16 v16bf;
typedef __attribute__((ext_vector_type(8)))  float  v8f;

#define HW   16384   // 128*128
#define CH   64
#define H_   128
#define W_   128
#define B_   8

__device__ __forceinline__ unsigned short f2bf(float f) {
    unsigned int u = __float_as_uint(f);
    unsigned int r = u + 0x7FFFu + ((u >> 16) & 1u);   // round-to-nearest-even
    return (unsigned short)(r >> 16);
}

// ---------------------------------------------------------------------------
// K0: fused  x=g+t -> 1x1 reduce(64->16) -> BN+ReLU -> 1x1 span(16->18)+bias
//     -> involution (groups=2, 3x3, per-pixel kernel) -> bf16 output
// one thread per pixel (b,h,w)
// ---------------------------------------------------------------------------
__global__ __launch_bounds__(256)
void k_kernelgen_involution(const float* __restrict__ guide,
                            const float* __restrict__ target,
                            const float* __restrict__ w_reduce,  // [16][64]
                            const float* __restrict__ bn1_g, const float* __restrict__ bn1_b,
                            const float* __restrict__ bn1_m, const float* __restrict__ bn1_v,
                            const float* __restrict__ w_span,    // [18][16]
                            const float* __restrict__ b_span,    // [18]
                            unsigned short* __restrict__ inv_bf16)
{
    int idx = blockIdx.x * 256 + threadIdx.x;          // pixel over B*H*W
    if (idx >= B_ * HW) return;
    int w = idx & (W_ - 1);
    int h = (idx >> 7) & (H_ - 1);
    int b = idx >> 14;
    int pbase = b * CH * HW + h * W_ + w;

    // 1x1 reduce accumulated channel-by-channel (w_reduce reads are wave-uniform)
    float xr[16];
    #pragma unroll
    for (int r = 0; r < 16; ++r) xr[r] = 0.f;
    for (int c = 0; c < CH; ++c) {
        float x = guide[pbase + c * HW] + target[pbase + c * HW];
        #pragma unroll
        for (int r = 0; r < 16; ++r) xr[r] += w_reduce[r * CH + c] * x;
    }
    // BN1 + ReLU
    #pragma unroll
    for (int r = 0; r < 16; ++r) {
        float sc = bn1_g[r] * rsqrtf(bn1_v[r] + 1e-5f);
        float y  = xr[r] * sc + (bn1_b[r] - bn1_m[r] * sc);
        xr[r] = y > 0.f ? y : 0.f;
    }
    // span conv -> 18 per-pixel kernel taps
    float kern[18];
    #pragma unroll
    for (int k = 0; k < 18; ++k) {
        float a = b_span[k];
        #pragma unroll
        for (int r = 0; r < 16; ++r) a += w_span[k * 16 + r] * xr[r];
        kern[k] = a;
    }
    // involution over all 64 channels (group = ch>>5), zero-padded 3x3 window
    for (int c = 0; c < CH; ++c) {
        const float* kv = &kern[(c >> 5) * 9];
        const float* tc = target + b * CH * HW + c * HW;
        float acc = 0.f;
        #pragma unroll
        for (int kh = 0; kh < 3; ++kh) {
            int hh = h + kh - 1;
            if (hh < 0 || hh >= H_) continue;
            #pragma unroll
            for (int kw = 0; kw < 3; ++kw) {
                int ww = w + kw - 1;
                if (ww < 0 || ww >= W_) continue;
                acc += kv[kh * 3 + kw] * tc[hh * W_ + ww];
            }
        }
        inv_bf16[b * CH * HW + c * HW + h * W_ + w] = f2bf(acc);
    }
}

// ---------------------------------------------------------------------------
// K1: repack 3x3 conv weights  f32 [co][ci][kh][kw] -> bf16 [p][co][ci]
// ---------------------------------------------------------------------------
__global__ __launch_bounds__(256)
void k_repack_w(const float* __restrict__ wsrc, unsigned short* __restrict__ wdst)
{
    int i = blockIdx.x * 256 + threadIdx.x;            // (p*64 + co)*64 + ci
    if (i >= 9 * CH * CH) return;
    int ci = i & 63;
    int t  = i >> 6;
    int co = t & 63;
    int p  = t >> 6;
    wdst[i] = f2bf(wsrc[(co * CH + ci) * 9 + p]);
}

// ---------------------------------------------------------------------------
// K2: implicit-GEMM 3x3 conv 64->64 via v_wmma_f32_16x16x32_bf16, + BN + ReLU
// block = 128 threads (4 waves); tile = one full row: 128 pixels x 64 out ch.
// wave wv owns TWO M-subtiles (pixels 32*wv .. 32*wv+31), loops 4 N-tiles of
// 16 co, reusing each B fragment for 2 WMMAs.
// K = 576 ordered as k = p*64 + ci  (p = kh*3+kw) -> 18 chunks of K=32.
// ---------------------------------------------------------------------------
union AFrag { v16bf v; unsigned int u[8]; };
union BFrag { v16bf v; uint4 q[2]; };

__global__ __launch_bounds__(128)
void k_conv_wmma(const unsigned short* __restrict__ inv,   // bf16 [b][ci][h][w]
                 const unsigned short* __restrict__ wts,   // bf16 [p][co][ci]
                 const float* __restrict__ bn3_g, const float* __restrict__ bn3_b,
                 const float* __restrict__ bn3_m, const float* __restrict__ bn3_v,
                 float* __restrict__ out)
{
    __shared__ __align__(16) unsigned short s_x[3 * 130 * CH];  // [r][c][ci], 49.9 KB

    int blk = blockIdx.x;             // 1024 blocks: b(8) * h(128)
    int h   = blk & (H_ - 1);
    int b   = blk >> 7;
    int tid = threadIdx.x;

    // warm the (L2-resident) weight array for this block's B-fragment gathers
    __builtin_prefetch((const char*)wts + tid * 576, 0, 0);

    // --- stage zero-padded input halo row to LDS (coalesced along w) ---
    const int E = 3 * CH * 130;
    for (int e = tid; e < E; e += 128) {
        int c  = e % 130;             // LDS col 0..129  <->  gw = c-1
        int t  = e / 130;
        int ci = t & 63;
        int r  = t >> 6;              // 0..2  <->  gh = h-1+r
        int gh = h - 1 + r, gw = c - 1;
        unsigned short val = 0;
        if (gh >= 0 && gh < H_ && gw >= 0 && gw < W_)
            val = inv[((b * CH + ci) * H_ + gh) * W_ + gw];
        s_x[(r * 130 + c) * CH + ci] = val;
    }
    __syncthreads();

    int lane  = tid & 31;
    int wv    = tid >> 5;                       // wave id -> pixels 32*wv..+31
    int mrow  = lane & 15;                      // A-matrix row for this lane
    int kbase = (lane < 16) ? 0 : 8;            // A layout K base (ISA 7.12.2)
    int koff  = (lane < 16) ? 0 : 16;           // B layout K offset
    int n     = lane & 15;                      // B/D column

    v8f zero = {0.f, 0.f, 0.f, 0.f, 0.f, 0.f, 0.f, 0.f};
    v8f acc[2][4];
    #pragma unroll
    for (int mt = 0; mt < 2; ++mt)
        #pragma unroll
        for (int nt = 0; nt < 4; ++nt) acc[mt][nt] = zero;

    const unsigned int* sx32 = (const unsigned int*)s_x;
    const uint4*        wq   = (const uint4*)wts;

    for (int t = 0; t < 18; ++t) {              // K chunks of 32
        int p   = t >> 1;
        int cib = (t & 1) << 5;                 // ci base: 0 or 32
        int r   = p / 3;
        int c0  = wv * 32 + mrow + (p % 3);     // LDS col of M-subtile 0

        // two A fragments (M-subtiles), 8 x b32 each from LDS
        AFrag a0, a1;
        int xb0 = (r * 130 + c0) * CH + cib;          // bf16 index, even
        int xb1 = xb0 + 16 * CH;                      // +16 pixels
        #pragma unroll
        for (int vv = 0; vv < 8; ++vv) {
            int lk = koff + 2 * vv;
            a0.u[vv] = sx32[(xb0 + lk) >> 1];
            a1.u[vv] = sx32[(xb1 + lk) >> 1];
        }

        #pragma unroll
        for (int nt = 0; nt < 4; ++nt) {
            int co = nt * 16 + n;
            int wi = (p * CH + co) * CH + cib + koff; // bf16 index, mult of 16
            BFrag bf;
            bf.q[0] = wq[(wi >> 3) + 0];
            bf.q[1] = wq[(wi >> 3) + 1];
            acc[0][nt] = __builtin_amdgcn_wmma_f32_16x16x32_bf16(
                false, a0.v, false, bf.v, (short)0, acc[0][nt], false, false);
            acc[1][nt] = __builtin_amdgcn_wmma_f32_16x16x32_bf16(
                false, a1.v, false, bf.v, (short)0, acc[1][nt], false, false);
        }
    }

    // --- BN3 + ReLU epilogue, scatter f32 stores per D-layout ---
    #pragma unroll
    for (int nt = 0; nt < 4; ++nt) {
        int co = nt * 16 + n;
        float sc = bn3_g[co] * rsqrtf(bn3_v[co] + 1e-5f);
        float sh = bn3_b[co] - bn3_m[co] * sc;
        #pragma unroll
        for (int mt = 0; mt < 2; ++mt) {
            #pragma unroll
            for (int r = 0; r < 8; ++r) {
                int mm   = r + ((lane >> 4) << 3);    // D row per ISA layout
                int wpix = wv * 32 + mt * 16 + mm;
                float v  = acc[mt][nt][r] * sc + sh;
                v = v > 0.f ? v : 0.f;
                out[((b * CH + co) * H_ + h) * W_ + wpix] = v;
            }
        }
    }
}

// ---------------------------------------------------------------------------
extern "C" void kernel_launch(void* const* d_in, const int* in_sizes, int n_in,
                              void* d_out, int out_size, void* d_ws, size_t ws_size,
                              hipStream_t stream) {
    const float* guide    = (const float*)d_in[0];
    const float* target   = (const float*)d_in[1];
    const float* w_reduce = (const float*)d_in[2];
    const float* bn1_g    = (const float*)d_in[3];
    const float* bn1_b    = (const float*)d_in[4];
    const float* bn1_m    = (const float*)d_in[5];
    const float* bn1_v    = (const float*)d_in[6];
    const float* w_span   = (const float*)d_in[7];
    const float* b_span   = (const float*)d_in[8];
    const float* w_bconv  = (const float*)d_in[9];
    const float* bn3_g    = (const float*)d_in[10];
    const float* bn3_b    = (const float*)d_in[11];
    const float* bn3_m    = (const float*)d_in[12];
    const float* bn3_v    = (const float*)d_in[13];
    float* out = (float*)d_out;

    unsigned short* wsW = (unsigned short*)d_ws;                        // 73,728 B
    unsigned short* inv = (unsigned short*)((char*)d_ws + (1 << 17));   // 16.78 MB

    // K1: weight repack
    k_repack_w<<<(9 * CH * CH + 255) / 256, 256, 0, stream>>>(w_bconv, wsW);

    // K0: fused kernel-generation + involution -> bf16
    k_kernelgen_involution<<<(B_ * HW + 255) / 256, 256, 0, stream>>>(
        guide, target, w_reduce, bn1_g, bn1_b, bn1_m, bn1_v,
        w_span, b_span, inv);

    // K2: WMMA 3x3 conv + BN + ReLU (one row per block)
    k_conv_wmma<<<B_ * H_, 128, 0, stream>>>(
        inv, wsW, bn3_g, bn3_b, bn3_m, bn3_v, out);
}
